// MoEBase_39316130628255
// MI455X (gfx1250) — compile-verified, matrix operand order checked by reading, not run on hardware
//
#include <hip/hip_runtime.h>
#include <hip/hip_bf16.h>

#define D_MODEL 1024
#define HIDDEN 2048
#define NE 8
#define TOPK 2
#define T_TOKENS 8192

#define BM 128
#define BN 64
#define BK 32
#define BKP 40   // padded LDS K-stride in elements: 80 bytes, multiple of 16

#if defined(__gfx1250__) && __has_builtin(__builtin_amdgcn_global_load_async_to_lds_b128)
#define HAS_ASYNC_LDS 1
#else
#define HAS_ASYNC_LDS 0
#endif

typedef __bf16 bf16_t;
typedef bf16_t v16bf __attribute__((ext_vector_type(16)));
typedef float  v8f   __attribute__((ext_vector_type(8)));
typedef int    v4i   __attribute__((ext_vector_type(4)));

// f32 -> bf16 with round-half-up (1 VALU op + shift; vs ~4 for exact RNE).
__device__ __forceinline__ unsigned short f2bf(float x) {
  return (unsigned short)((__float_as_uint(x) + 0x8000u) >> 16);
}

// Two f32 -> packed bf16x2: two adds + one v_perm_b32.
__device__ __forceinline__ unsigned int pack2(float lo, float hi) {
  const unsigned int ul = __float_as_uint(lo) + 0x8000u;
  const unsigned int uh = __float_as_uint(hi) + 0x8000u;
  return __builtin_amdgcn_perm(uh, ul, 0x07060302u);  // {uh[31:16], ul[31:16]}
}

__device__ __forceinline__ v8f zero8() {
  v8f z;
#pragma unroll
  for (int i = 0; i < 8; ++i) z[i] = 0.0f;
  return z;
}

// LDS-only workgroup barrier: avoids __syncthreads()'s full vmem drain
// (s_wait_loadcnt at the barrier would kill register-prefetch overlap).
__device__ __forceinline__ void barrier_lds() {
  __builtin_amdgcn_fence(__ATOMIC_RELEASE, "workgroup", "local");
  __builtin_amdgcn_s_barrier();
  __builtin_amdgcn_fence(__ATOMIC_ACQUIRE, "workgroup", "local");
}

// 16x32 bf16 WMMA fragment from an LDS row (K contiguous): two ds_load_b128.
// Lane (l&15) selects row/col; kbase=(l>>4)*8 per ISA 16-bit operand layout.
__device__ __forceinline__ v16bf load_frag(const unsigned short* rowbase, int kbase) {
  union { uint4 u[2]; v16bf v; } f;
  f.u[0] = *(const uint4*)(rowbase + kbase);       // K = kbase .. kbase+7
  f.u[1] = *(const uint4*)(rowbase + kbase + 16);  // K = kbase+16 .. kbase+23
  return f.v;
}

// ---------------------------------------------------------------------------
// Router: one wave per token. logits = x@wr, softmax, top-2, atomic ranks.
// ---------------------------------------------------------------------------
__global__ __launch_bounds__(256) void router_kernel(
    const float* __restrict__ X, const float* __restrict__ WR,
    int* __restrict__ counts, int* __restrict__ eids,
    int* __restrict__ ranks, float* __restrict__ wtops) {
  __shared__ float wrs[D_MODEL * NE];
  const int tid = threadIdx.x;
  for (int i = tid; i < D_MODEL * NE; i += 256) wrs[i] = WR[i];
  __syncthreads();

  const int lane = tid & 31;
  const int wv = tid >> 5;
  const int t = blockIdx.x * 8 + wv;

  float acc[NE];
#pragma unroll
  for (int e = 0; e < NE; ++e) acc[e] = 0.0f;

  for (int i = 0; i < D_MODEL / 128; ++i) {
    const int k = i * 128 + lane * 4;
    float4 xv = *(const float4*)(X + (size_t)t * D_MODEL + k);
    float xs[4] = {xv.x, xv.y, xv.z, xv.w};
#pragma unroll
    for (int j = 0; j < 4; ++j) {
      const float* w = &wrs[(k + j) * NE];
#pragma unroll
      for (int e = 0; e < NE; ++e) acc[e] += xs[j] * w[e];
    }
  }
#pragma unroll
  for (int off = 16; off > 0; off >>= 1)
#pragma unroll
    for (int e = 0; e < NE; ++e) acc[e] += __shfl_xor(acc[e], off);

  if (lane == 0) {
    float mx = acc[0];
#pragma unroll
    for (int e = 1; e < NE; ++e) mx = fmaxf(mx, acc[e]);
    float p[NE], sum = 0.0f;
#pragma unroll
    for (int e = 0; e < NE; ++e) { p[e] = __expf(acc[e] - mx); sum += p[e]; }
    const float inv = 1.0f / sum;
#pragma unroll
    for (int e = 0; e < NE; ++e) p[e] *= inv;

    int prev = -1;
    for (int kk = 0; kk < TOPK; ++kk) {
      int best = 0; float bv = -1.0f;
      for (int e = 0; e < NE; ++e) {
        if (e == prev) continue;
        if (p[e] > bv) { bv = p[e]; best = e; }
      }
      const int r = atomicAdd(&counts[best], 1);
      eids[t * TOPK + kk] = best;
      ranks[t * TOPK + kk] = r;
      wtops[t * TOPK + kk] = bv;
      prev = best;
    }
  }
}

__global__ void zero_counts(int* counts) {
  if (threadIdx.x < NE) counts[threadIdx.x] = 0;
}

__global__ void scan_kernel(const int* __restrict__ counts, int* __restrict__ offs) {
  if (threadIdx.x == 0 && blockIdx.x == 0) {
    int s = 0;
    for (int e = 0; e < NE; ++e) { offs[e] = s; s += counts[e]; }
  }
}

__global__ __launch_bounds__(256) void scatter_kernel(
    const int* __restrict__ eids, const int* __restrict__ ranks,
    const float* __restrict__ wtops, const int* __restrict__ offs,
    int* __restrict__ tok, float* __restrict__ aw) {
  const int i = blockIdx.x * 256 + threadIdx.x;
  if (i >= T_TOKENS * TOPK) return;
  const int e = eids[i];
  const int row = offs[e] + ranks[i];
  tok[row] = i / TOPK;
  aw[row] = (2.0f / 3.0f) * wtops[i];
}

// ---------------------------------------------------------------------------
// GEMM 1: H = silu(X @ W1) * (X @ W3), f32 in -> bf16 out, optional gather.
// Block 128x64, K-step 32, 8 waves, wave tile 32x32 (2x2 WMMA, two acc sets).
// Double-buffered LDS, register-prefetched global loads, LDS-only barriers.
// ---------------------------------------------------------------------------
template <bool GATHER>
__global__ __launch_bounds__(256) void gemm_swiglu_kernel(
    const float* __restrict__ X, const float* __restrict__ W1b,
    const float* __restrict__ W3b, unsigned short* __restrict__ Hout,
    const int* __restrict__ tok_idx, const int* __restrict__ seg_off,
    const int* __restrict__ seg_cnt, int rows_total) {
  __shared__ __align__(16) unsigned short As[2][BM][BKP];
  __shared__ __align__(16) unsigned short B1s[2][BN][BKP];
  __shared__ __align__(16) unsigned short B3s[2][BN][BKP];

  const int tid = threadIdx.x;
  const int e = blockIdx.z;
  const int row0 = blockIdx.x * BM;
  const int n0 = blockIdx.y * BN;

  const float* W1 = W1b;
  const float* W3 = W3b;
  const int* idx = nullptr;
  int nrows, hbase;
  if (GATHER) {
    const int cnt = seg_cnt[e];
    if (row0 >= cnt) return;
    nrows = cnt - row0; if (nrows > BM) nrows = BM;
    const int off = seg_off[e];
    idx = tok_idx + off;
    hbase = off + row0;
    W1 += (size_t)e * D_MODEL * HIDDEN;
    W3 += (size_t)e * D_MODEL * HIDDEN;
  } else {
    nrows = rows_total - row0; if (nrows > BM) nrows = BM;
    hbase = row0;
  }

  const int lane = tid & 31;
  const int wv = tid >> 5;
  const int wm = wv & 3;             // 4 row groups of 32
  const int wn = wv >> 2;            // 2 col groups of 32
  const int lm = lane & 15;
  const int kbase = (lane >> 4) * 8;

  v8f acc1[2][2], acc3[2][2];
#pragma unroll
  for (int mi = 0; mi < 2; ++mi)
#pragma unroll
    for (int ni = 0; ni < 2; ++ni) { acc1[mi][ni] = zero8(); acc3[mi][ni] = zero8(); }

  float4 aReg[4];
  float4 b1Reg[2], b3Reg[2];

  // A: 128x32 tile, thread owns 4 float4 strips (rows clamped; excess rows unused).
  auto loadA = [&](int k0) {
#pragma unroll
    for (int i = 0; i < 4; ++i) {
      const int lin = tid + i * 256;
      int r = lin >> 3;
      if (r >= nrows) r = nrows - 1;
      const int c = (lin & 7) << 2;
      const int grow = GATHER ? idx[row0 + r] : (row0 + r);
      aReg[i] = *(const float4*)(X + (size_t)grow * D_MODEL + (k0 + c));
    }
  };
  // B: 32x64 tile per matrix; thread owns (n, k..k+3): 4 coalesced b32 loads.
  auto loadB = [&](int k0) {
#pragma unroll
    for (int i = 0; i < 2; ++i) {
      const int lin = tid + i * 256;
      const int n = lin & 63;
      const int kk = (lin >> 6) << 2;
      const float* p1 = W1 + (size_t)(k0 + kk) * HIDDEN + (n0 + n);
      const float* p3 = W3 + (size_t)(k0 + kk) * HIDDEN + (n0 + n);
      b1Reg[i] = make_float4(p1[0], p1[HIDDEN], p1[2 * HIDDEN], p1[3 * HIDDEN]);
      b3Reg[i] = make_float4(p3[0], p3[HIDDEN], p3[2 * HIDDEN], p3[3 * HIDDEN]);
    }
  };
  auto commit = [&](int buf) {
#pragma unroll
    for (int i = 0; i < 4; ++i) {
      const int lin = tid + i * 256;
      const int r = lin >> 3;
      const int c = (lin & 7) << 2;
      *(uint2*)(&As[buf][r][c]) =
          make_uint2(pack2(aReg[i].x, aReg[i].y), pack2(aReg[i].z, aReg[i].w));
    }
#pragma unroll
    for (int i = 0; i < 2; ++i) {
      const int lin = tid + i * 256;
      const int n = lin & 63;
      const int kk = (lin >> 6) << 2;
      *(uint2*)(&B1s[buf][n][kk]) =
          make_uint2(pack2(b1Reg[i].x, b1Reg[i].y), pack2(b1Reg[i].z, b1Reg[i].w));
      *(uint2*)(&B3s[buf][n][kk]) =
          make_uint2(pack2(b3Reg[i].x, b3Reg[i].y), pack2(b3Reg[i].z, b3Reg[i].w));
    }
  };
  auto compute = [&](int buf) {
    v16bf a[2], b1[2], b3[2];
#pragma unroll
    for (int mi = 0; mi < 2; ++mi)
      a[mi] = load_frag(&As[buf][wm * 32 + mi * 16 + lm][0], kbase);
#pragma unroll
    for (int ni = 0; ni < 2; ++ni) {
      b1[ni] = load_frag(&B1s[buf][wn * 32 + ni * 16 + lm][0], kbase);
      b3[ni] = load_frag(&B3s[buf][wn * 32 + ni * 16 + lm][0], kbase);
    }
#pragma unroll
    for (int mi = 0; mi < 2; ++mi)
#pragma unroll
      for (int ni = 0; ni < 2; ++ni) {
        acc1[mi][ni] = __builtin_amdgcn_wmma_f32_16x16x32_bf16(
            false, a[mi], false, b1[ni], (short)0, acc1[mi][ni], false, false);
        acc3[mi][ni] = __builtin_amdgcn_wmma_f32_16x16x32_bf16(
            false, a[mi], false, b3[ni], (short)0, acc3[mi][ni], false, false);
      }
  };

  loadA(0); loadB(0); commit(0);
  int cur = 0;
  for (int k0 = BK; k0 < D_MODEL; k0 += BK) {
    loadA(k0); loadB(k0);       // global loads stay in flight across barrier
    barrier_lds();              // cur buffer fully staged (dscnt only)
    compute(cur);
    commit(cur ^ 1);            // convert + store into the other buffer
    cur ^= 1;
  }
  barrier_lds();
  compute(cur);

  // Epilogue: silu(a1)*a3 -> bf16. C layout: VGPR i -> M = i + 8*(lane>>4), N = lane&15.
  const int mloc = (lane >> 4) * 8;
  const int ncol = lane & 15;
#pragma unroll
  for (int mi = 0; mi < 2; ++mi)
#pragma unroll
    for (int ni = 0; ni < 2; ++ni)
#pragma unroll
      for (int i = 0; i < 8; ++i) {
        const int m = wm * 32 + mi * 16 + mloc + i;
        if (m < nrows) {
          const int nn = n0 + wn * 32 + ni * 16 + ncol;
          const float v1 = acc1[mi][ni][i];
          const float v3 = acc3[mi][ni][i];
          const float h = (v1 / (1.0f + __expf(-v1))) * v3;
          Hout[(size_t)(hbase + m) * HIDDEN + nn] = f2bf(h);
        }
      }
}

// ---------------------------------------------------------------------------
// GEMM 2: Y = H(bf16) @ W2(f32->bf16). ATOMIC: scatter-add weight*y into out.
// A tile is a raw bf16 copy -> async global->LDS loads (b128) when available.
// ---------------------------------------------------------------------------
template <bool ATOMIC>
__global__ __launch_bounds__(256) void gemm_out_kernel(
    const unsigned short* __restrict__ Hin, const float* __restrict__ W2b,
    float* __restrict__ Out, const int* __restrict__ tok_idx,
    const float* __restrict__ awgt, const int* __restrict__ seg_off,
    const int* __restrict__ seg_cnt, int rows_total, float scale) {
  __shared__ __align__(16) unsigned short As[2][BM][BKP];
  __shared__ __align__(16) unsigned short Bs[2][BN][BKP];

  const int tid = threadIdx.x;
  const int e = blockIdx.z;
  const int row0 = blockIdx.x * BM;
  const int n0 = blockIdx.y * BN;

  const float* W2 = W2b;
  int nrows, abase;
  if (ATOMIC) {
    const int cnt = seg_cnt[e];
    if (row0 >= cnt) return;
    nrows = cnt - row0; if (nrows > BM) nrows = BM;
    abase = seg_off[e] + row0;
    W2 += (size_t)e * HIDDEN * D_MODEL;
  } else {
    nrows = rows_total - row0; if (nrows > BM) nrows = BM;
    abase = row0;
  }

  const int lane = tid & 31;
  const int wv = tid >> 5;
  const int wm = wv & 3;
  const int wn = wv >> 2;
  const int lm = lane & 15;
  const int kbase = (lane >> 4) * 8;

  v8f acc[2][2];
#pragma unroll
  for (int mi = 0; mi < 2; ++mi)
#pragma unroll
    for (int ni = 0; ni < 2; ++ni) acc[mi][ni] = zero8();

  float4 bReg[2];
#if !HAS_ASYNC_LDS
  uint4 aReg[2];
#endif

#if HAS_ASYNC_LDS
  // Async A staging: 2 x b128 per thread, wave-uniform issue (rows clamped).
  auto issueA = [&](int buf, int k0) {
#pragma unroll
    for (int i = 0; i < 2; ++i) {
      const int lin = tid + i * 256;
      int r = lin >> 2;
      if (r >= nrows) r = nrows - 1;
      const int c8 = (lin & 3) << 3;
      __builtin_amdgcn_global_load_async_to_lds_b128(
          (__attribute__((address_space(1))) v4i*)(Hin + (size_t)(abase + r) * HIDDEN + k0 + c8),
          (__attribute__((address_space(3))) v4i*)(&As[buf][lin >> 2][c8]),
          0, 0);
    }
  };
#else
  auto loadA = [&](int k0) {
#pragma unroll
    for (int i = 0; i < 2; ++i) {
      const int lin = tid + i * 256;
      int r = lin >> 2;
      if (r >= nrows) r = nrows - 1;
      const int c8 = (lin & 3) << 3;
      aReg[i] = *(const uint4*)(Hin + (size_t)(abase + r) * HIDDEN + k0 + c8);
    }
  };
  auto commitA = [&](int buf) {
#pragma unroll
    for (int i = 0; i < 2; ++i) {
      const int lin = tid + i * 256;
      const int r = lin >> 2;
      const int c8 = (lin & 3) << 3;
      *(uint4*)(&As[buf][r][c8]) = aReg[i];
    }
  };
#endif
  auto loadB = [&](int k0) {
#pragma unroll
    for (int i = 0; i < 2; ++i) {
      const int lin = tid + i * 256;
      const int n = lin & 63;
      const int kk = (lin >> 6) << 2;
      const float* p = W2 + (size_t)(k0 + kk) * D_MODEL + (n0 + n);
      bReg[i] = make_float4(p[0], p[D_MODEL], p[2 * D_MODEL], p[3 * D_MODEL]);
    }
  };
  auto commitB = [&](int buf) {
#pragma unroll
    for (int i = 0; i < 2; ++i) {
      const int lin = tid + i * 256;
      const int n = lin & 63;
      const int kk = (lin >> 6) << 2;
      *(uint2*)(&Bs[buf][n][kk]) =
          make_uint2(pack2(bReg[i].x, bReg[i].y), pack2(bReg[i].z, bReg[i].w));
    }
  };
  auto compute = [&](int buf) {
    v16bf a[2], b[2];
#pragma unroll
    for (int mi = 0; mi < 2; ++mi)
      a[mi] = load_frag(&As[buf][wm * 32 + mi * 16 + lm][0], kbase);
#pragma unroll
    for (int ni = 0; ni < 2; ++ni)
      b[ni] = load_frag(&Bs[buf][wn * 32 + ni * 16 + lm][0], kbase);
#pragma unroll
    for (int mi = 0; mi < 2; ++mi)
#pragma unroll
      for (int ni = 0; ni < 2; ++ni)
        acc[mi][ni] = __builtin_amdgcn_wmma_f32_16x16x32_bf16(
            false, a[mi], false, b[ni], (short)0, acc[mi][ni], false, false);
  };

#if HAS_ASYNC_LDS
  issueA(0, 0);
#else
  loadA(0);
#endif
  loadB(0);
#if !HAS_ASYNC_LDS
  commitA(0);
#endif
  commitB(0);
  int cur = 0;
  for (int k0 = BK; k0 < HIDDEN; k0 += BK) {
#if !HAS_ASYNC_LDS
    loadA(k0);
#endif
    loadB(k0);
#if HAS_ASYNC_LDS
    asm volatile("s_wait_asynccnt 0x0" ::: "memory");  // cur-buf async writes done
#endif
    barrier_lds();
#if HAS_ASYNC_LDS
    issueA(cur ^ 1, k0);   // overlaps with compute(cur); safe: next-buf reads done
#endif
    compute(cur);
#if !HAS_ASYNC_LDS
    commitA(cur ^ 1);
#endif
    commitB(cur ^ 1);
    cur ^= 1;
  }
#if HAS_ASYNC_LDS
  asm volatile("s_wait_asynccnt 0x0" ::: "memory");
#endif
  barrier_lds();
  compute(cur);

  const int mloc = (lane >> 4) * 8;
  const int ncol = lane & 15;
#pragma unroll
  for (int mi = 0; mi < 2; ++mi)
#pragma unroll
    for (int ni = 0; ni < 2; ++ni)
#pragma unroll
      for (int i = 0; i < 8; ++i) {
        const int m = wm * 32 + mi * 16 + mloc + i;
        if (m < nrows) {
          const int col = n0 + wn * 32 + ni * 16 + ncol;
          const float v = acc[mi][ni][i];
          if (ATOMIC) {
            const int arow = abase + m;
            const int t = tok_idx[arow];
            atomicAdd(&Out[(size_t)t * D_MODEL + col], awgt[arow] * v);
          } else {
            Out[(size_t)(abase + m) * D_MODEL + col] = scale * v;
          }
        }
      }
}

// ---------------------------------------------------------------------------
extern "C" void kernel_launch(void* const* d_in, const int* in_sizes, int n_in,
                              void* d_out, int out_size, void* d_ws, size_t ws_size,
                              hipStream_t stream) {
  (void)in_sizes; (void)n_in; (void)out_size; (void)ws_size;
  const float* x   = (const float*)d_in[0];
  const float* wr  = (const float*)d_in[1];
  const float* w1  = (const float*)d_in[2];
  const float* w3  = (const float*)d_in[3];
  const float* w2  = (const float*)d_in[4];
  const float* ws1 = (const float*)d_in[5];
  const float* ws3 = (const float*)d_in[6];
  const float* ws2 = (const float*)d_in[7];
  float* out = (float*)d_out;

  unsigned char* ws = (unsigned char*)d_ws;
  int*   counts = (int*)(ws + 0);
  int*   offs   = (int*)(ws + 64);
  int*   eids   = (int*)(ws + 128);
  int*   ranks  = eids + T_TOKENS * TOPK;
  float* wtops  = (float*)(ranks + T_TOKENS * TOPK);
  int*   tok    = (int*)(wtops + T_TOKENS * TOPK);
  float* aw     = (float*)(tok + T_TOKENS * TOPK);
  unsigned short* Hexp = (unsigned short*)(ws + 524288);                          // 64 MB
  unsigned short* Hsh  = (unsigned short*)(ws + 524288 +
                           (size_t)T_TOKENS * TOPK * HIDDEN * sizeof(unsigned short)); // 32 MB

  zero_counts<<<1, 32, 0, stream>>>(counts);
  router_kernel<<<T_TOKENS / 8, 256, 0, stream>>>(x, wr, counts, eids, ranks, wtops);
  scan_kernel<<<1, 1, 0, stream>>>(counts, offs);
  scatter_kernel<<<(T_TOKENS * TOPK + 255) / 256, 256, 0, stream>>>(eids, ranks, wtops, offs, tok, aw);

  // Shared MLP: Hs = silu(x@ws1)*(x@ws3); out = (1/3) * Hs@ws2 (initializes out)
  dim3 gs1(T_TOKENS / BM, HIDDEN / BN, 1);
  gemm_swiglu_kernel<false><<<gs1, 256, 0, stream>>>(x, ws1, ws3, Hsh,
                                                     nullptr, nullptr, nullptr, T_TOKENS);
  dim3 gs2(T_TOKENS / BM, D_MODEL / BN, 1);
  gemm_out_kernel<false><<<gs2, 256, 0, stream>>>(Hsh, ws2, out, nullptr, nullptr,
                                                  nullptr, nullptr, T_TOKENS, 1.0f / 3.0f);

  // Experts (grouped, gathered): H = silu(Xg@w1)*(Xg@w3); out += (2/3)*w * H@w2
  dim3 ge1(T_TOKENS / BM, HIDDEN / BN, NE);
  gemm_swiglu_kernel<true><<<ge1, 256, 0, stream>>>(x, w1, w3, Hexp,
                                                    tok, offs, counts, 0);
  dim3 ge2(T_TOKENS / BM, D_MODEL / BN, NE);
  gemm_out_kernel<true><<<ge2, 256, 0, stream>>>(Hexp, w2, out, tok, aw,
                                                 offs, counts, 0, 1.0f);
}
